// MyKALSTM_73065983640223
// MI455X (gfx1250) — compile-verified, matrix operand order checked by reading, not run on hardware
//
#include <hip/hip_runtime.h>
#include <hip/hip_bf16.h>

typedef __attribute__((ext_vector_type(16))) _Float16 v16h;
typedef __attribute__((ext_vector_type(8)))  _Float16 v8h;
typedef __attribute__((ext_vector_type(8)))  float    v8f;

// ---------------------------------------------------------------- constants
constexpr int cB = 32, cP = 12, cQ = 12, cD = 64;
constexpr int cN = 1024, cC = 1024;           // nodes / cells (H*W = 32*32)
constexpr int cPQ = 24;                        // P+Q

union HV16 { v16h v; v8h h[2]; };
union HPK  { _Float16 h[2]; unsigned u; };

// LDS byte offset of a __shared__ object: generic LDS pointers carry the
// LDS offset in their low 32 bits (ISA 10.2 aperture mapping).
__device__ inline unsigned lds_off(const void* p)
{
    return (unsigned)(unsigned long long)p;
}

// ======================================================================
//  Generic batched WMMA GEMM:  D = act(alpha * A@B + beta * Cres + bias)
//  A: (M,K) f16 row-major, batch stride sA (0 => shared across batch)
//  B: (K,Nc) f16 row-major, batch stride sB
//  Workgroup: 256 threads = 8 wave32; tile 128(M) x 64(N); K panel = 64.
//  Each wave: 32x32 region; 8 x v_wmma_f32_16x16x32_f16 per panel.
//  Double-buffered software pipeline: panel p+1 is staged (A via
//  GLOBAL_LOAD_ASYNC_TO_LDS_B128 / ASYNCcnt, B via coalesced b128 loads +
//  packed-b32 transpose scatter) while panel p computes; one barrier per
//  panel. LDS tiles use stride 72 halves (144B): 16B alignment for the
//  ds_load_b128 fragments with bank skew.
// ======================================================================
__global__ __launch_bounds__(256)
void gk_wmma_gemm(const _Float16* __restrict__ A, long sA,
                  const _Float16* __restrict__ Bm, long sB,
                  const float* Cres, long sC, float beta,
                  const float* __restrict__ bias,
                  float* Dout, long sD,
                  _Float16* Dh, long sDh,
                  int M, int K, int Nc, float alpha, int act)
{
    constexpr int LDA = 72;             // padded row stride (halves)
    constexpr int LDB = 72;
    __shared__ alignas(16) _Float16 As[2][128 * LDA];   // [row][k]
    __shared__ alignas(16) _Float16 Bs[2][64 * LDB];    // [col][k] (transposed)

    const int tid   = threadIdx.x;
    const int lane  = tid & 31;
    const int wave  = tid >> 5;
    const int wm    = wave & 3;          // 0..3 : M sub-tile
    const int wn    = wave >> 2;         // 0..1 : N sub-tile
    const int lh    = lane & 15;
    const int khalf = (lane >> 4) << 3;  // lanes 16-31 take K+8 halves
    const int blockM = blockIdx.y * 128;
    const int blockN = blockIdx.x * 64;
    const long z = blockIdx.z;

    const _Float16* Ab = A  + z * sA;
    const _Float16* Bb = Bm + z * sB;

    // interior block (common case for this model) -> pure vector staging
    const bool interior = (blockM + 128 <= M) && (blockN + 64 <= Nc) &&
                          ((K & 63) == 0) && ((Nc & 7) == 0);

    v8f acc[2][2] = {};

    // ---- stage one 64-wide K panel into buffer `buf` ----
    auto stage = [&](int buf, int kb) {
        _Float16* Asb = As[buf];
        _Float16* Bsb = Bs[buf];
        if (interior) {
            // A: 128x64 tile, 64B per thread via async-to-LDS (ASYNCcnt)
            {
                int r  = tid >> 1;               // 0..127
                int c0 = (tid & 1) << 5;         // 0 or 32 (halves)
                const _Float16* ga = Ab + (long)(blockM + r) * K + kb + c0;
                unsigned la = lds_off(&Asb[r * LDA + c0]);
                asm volatile("global_load_async_to_lds_b128 %0, %1, off"
                             :: "v"(la), "v"(ga) : "memory");
                asm volatile("global_load_async_to_lds_b128 %0, %1, off offset:16"
                             :: "v"(la), "v"(ga) : "memory");
                asm volatile("global_load_async_to_lds_b128 %0, %1, off offset:32"
                             :: "v"(la), "v"(ga) : "memory");
                asm volatile("global_load_async_to_lds_b128 %0, %1, off offset:48"
                             :: "v"(la), "v"(ga) : "memory");
            }
            // B: two adjacent k-rows, packed-b32 transpose scatter
            {
                int t = tid >> 3;                // k-row pair 0..31
                int c = (tid & 7) << 3;          // col base 0..56
                const _Float16* g0 = Bb + (long)(kb + 2 * t) * Nc + blockN + c;
                v8h b0 = *reinterpret_cast<const v8h*>(g0);
                v8h b1 = *reinterpret_cast<const v8h*>(g0 + Nc);
#pragma unroll
                for (int j = 0; j < 8; ++j) {
                    HPK pk;
                    pk.h[0] = b0[j];
                    pk.h[1] = b1[j];
                    *reinterpret_cast<unsigned*>(&Bsb[(c + j) * LDB + 2 * t]) = pk.u;
                }
            }
            // warm L2 one panel further ahead (gfx1250 global_prefetch)
            if (kb + 64 < K)
                __builtin_prefetch(&Ab[(long)(blockM + (tid >> 1)) * K + kb + 64], 0, 1);
        } else {
            // cold path: guarded, zero-padded scalar staging
            for (int i = tid; i < 128 * 64; i += 256) {
                int r = i >> 6, c = i & 63;
                int gr = blockM + r, gk = kb + c;
                Asb[r * LDA + c] =
                    (gr < M && gk < K) ? Ab[(long)gr * K + gk] : (_Float16)0.f;
            }
            for (int i = tid; i < 64 * 64; i += 256) {
                int c = i >> 6, k = i & 63;      // Bs[c][k]
                int gk = kb + k, gn = blockN + c;
                Bsb[c * LDB + k] =
                    (gk < K && gn < Nc) ? Bb[(long)gk * Nc + gn] : (_Float16)0.f;
            }
        }
    };

    // ---- 2 k-steps x 2x2 WMMA on buffer `buf` ----
    auto compute = [&](int buf) {
        const _Float16* Asb = As[buf];
        const _Float16* Bsb = Bs[buf];
#pragma unroll
        for (int ks = 0; ks < 64; ks += 32) {
            HV16 af[2], bf[2];
#pragma unroll
            for (int mt = 0; mt < 2; ++mt) {
                int row = wm * 32 + mt * 16 + lh;
                af[mt].h[0] = *reinterpret_cast<const v8h*>(&Asb[row * LDA + ks + khalf]);
                af[mt].h[1] = *reinterpret_cast<const v8h*>(&Asb[row * LDA + ks + 16 + khalf]);
            }
#pragma unroll
            for (int nt = 0; nt < 2; ++nt) {
                int col = wn * 32 + nt * 16 + lh;
                bf[nt].h[0] = *reinterpret_cast<const v8h*>(&Bsb[col * LDB + ks + khalf]);
                bf[nt].h[1] = *reinterpret_cast<const v8h*>(&Bsb[col * LDB + ks + 16 + khalf]);
            }
#pragma unroll
            for (int mt = 0; mt < 2; ++mt)
#pragma unroll
                for (int nt = 0; nt < 2; ++nt)
                    acc[mt][nt] = __builtin_amdgcn_wmma_f32_16x16x32_f16(
                        false, af[mt].v, false, bf[nt].v, (short)0, acc[mt][nt],
                        false, false);
        }
    };

    // ---- software pipeline: one barrier per panel ----
    stage(0, 0);
    if (interior) asm volatile("s_wait_asynccnt 0x0" ::: "memory");
    __syncthreads();

    const int npan = (K + 63) >> 6;
    for (int p = 0; p < npan; ++p) {
        const int cur = p & 1;
        const bool more = (p + 1) < npan;
        if (more) stage(cur ^ 1, (p + 1) << 6);
        compute(cur);
        if (more) {
            if (interior) asm volatile("s_wait_asynccnt 0x0" ::: "memory");
            __syncthreads();
        }
    }

    // ---- epilogue: alpha/beta residual, bias, activation, f32 + f16 out ----
#pragma unroll
    for (int mt = 0; mt < 2; ++mt)
        for (int nt = 0; nt < 2; ++nt)
#pragma unroll
            for (int r = 0; r < 8; ++r) {
                int row = blockM + wm * 32 + mt * 16 + r + ((lane >= 16) ? 8 : 0);
                int col = blockN + wn * 32 + nt * 16 + lh;
                if (row < M && col < Nc) {
                    float v = alpha * acc[mt][nt][r];
                    if (Cres) v += beta * Cres[z * sC + (long)row * Nc + col];
                    if (bias) v += bias[col];
                    if      (act == 1) v = fmaxf(v, 0.f);
                    else if (act == 2) v = 1.f / (1.f + __expf(-v));
                    else if (act == 3) v = tanhf(v);
                    Dout[z * sD + (long)row * Nc + col] = v;
                    if (Dh) Dh[z * sDh + (long)row * Nc + col] = (_Float16)v;
                }
            }
}

// ======================================================================
//  Elementwise / setup kernels
// ======================================================================
__global__ void gk_zero(float* p, long n)
{
    long i = (long)blockIdx.x * 256 + threadIdx.x;
    if (i < n) p[i] = 0.f;
}

__global__ void gk_cast16(const float* __restrict__ s, _Float16* __restrict__ d, long n)
{
    long i = (long)blockIdx.x * 256 + threadIdx.x;
    if (i < n) d[i] = (_Float16)s[i];
}

// symmetric degree: d[n] = sum_m max(adj[n,m], adj[m,n])
__global__ void gk_degree(const float* __restrict__ adj, float* __restrict__ deg)
{
    __shared__ float red[256];
    int n = blockIdx.x;
    float s = 0.f;
    for (int m = threadIdx.x; m < cN; m += 256)
        s += fmaxf(adj[(long)n * cN + m], adj[(long)m * cN + n]);
    red[threadIdx.x] = s;
    __syncthreads();
    for (int o = 128; o > 0; o >>= 1) {
        if (threadIdx.x < o) red[threadIdx.x] += red[threadIdx.x + o];
        __syncthreads();
    }
    if (threadIdx.x == 0) deg[n] = red[0];
}

// support = L - I = -dinv[n] * a[n,m] * dinv[m]   (fp16 output, L2-resident)
__global__ void gk_support(const float* __restrict__ adj, const float* __restrict__ deg,
                           _Float16* __restrict__ s16)
{
    long i = (long)blockIdx.x * 256 + threadIdx.x;
    if (i >= (long)cN * cN) return;
    int n = (int)(i >> 10), m = (int)(i & 1023);
    float a  = fmaxf(adj[(long)n * cN + m], adj[(long)m * cN + n]);
    float dn = deg[n], dm = deg[m];
    float di = dn > 0.f ? rsqrtf(fmaxf(dn, 1e-12f)) : 0.f;
    float dj = dm > 0.f ? rsqrtf(fmaxf(dm, 1e-12f)) : 0.f;
    s16[i] = (_Float16)(-di * a * dj);
}

// temporal embedding: te(b,p,:) = fc2(te_mlp, onehot(dow)‖onehot(tod))
// one-hot @ w1 == w1[dow] + w1[7+tod]
__global__ void gk_te_embed(const int* __restrict__ TE,
                            const float* __restrict__ w1, const float* __restrict__ b1,
                            const float* __restrict__ w2, const float* __restrict__ b2,
                            float* __restrict__ out /* (B,P,D) */)
{
    __shared__ float hid[cD];
    int bp = blockIdx.x;                 // b*P + p
    int b = bp / cP, p = bp % cP;
    int dow = TE[((long)b * cPQ + p) * 2 + 0];
    int tod = TE[((long)b * cPQ + p) * 2 + 1];
    int j = threadIdx.x;
    float h = w1[(long)dow * cD + j] + w1[(long)(7 + tod) * cD + j] + b1[j];
    hid[j] = fmaxf(h, 0.f);
    __syncthreads();
    float o = b2[j];
    for (int t = 0; t < cD; ++t) o += hid[t] * w2[t * cD + j];
    out[(long)bp * cD + j] = o;
}

// input embed (din=1 fc2) + SE + TE:  out[b,p,c,:] per block
__global__ void gk_in1_embed(const float* __restrict__ Zin /* (B,P,nodes) */,
                             const float* __restrict__ w1, const float* __restrict__ b1,
                             const float* __restrict__ w2, const float* __restrict__ b2,
                             const float* __restrict__ se /* (nodes,D) */,
                             const float* __restrict__ te /* (B,P,D) */,
                             float* __restrict__ out /* (B,P,nodes,D) */)
{
    __shared__ float hid[cD];
    long bpc = blockIdx.x;
    int d = threadIdx.x;
    long node = bpc & 1023;
    long bp   = bpc >> 10;
    float zv = Zin[bpc];
    hid[d] = fmaxf(zv * w1[d] + b1[d], 0.f);
    __syncthreads();
    float o = b2[d];
    for (int t = 0; t < cD; ++t) o += hid[t] * w2[t * cD + d];
    o += se[node * cD + d] + te[bp * cD + d];
    out[bpc * cD + d] = o;
}

// im2col for 3x3 SAME conv over (B,32,32,64); x indexed with batch stride/offset
__global__ void gk_im2col(const float* __restrict__ x, long bstride, long off,
                          _Float16* __restrict__ col /* (B*1024, 576) */)
{
    long i = (long)blockIdx.x * 256 + threadIdx.x;
    const long total = (long)cB * 1024 * 576;
    if (i >= total) return;
    long px = i / 576;
    int f   = (int)(i % 576);
    int b  = (int)(px >> 10);
    int y  = (int)((px >> 5) & 31);
    int xw = (int)(px & 31);
    int ky = f / 192, kx = (f / 64) % 3, d = f & 63;
    int yy = y + ky - 1, xx = xw + kx - 1;
    float v = 0.f;
    if (yy >= 0 && yy < 32 && xx >= 0 && xx < 32)
        v = x[off + (long)b * bstride + (long)(yy * 32 + xx) * cD + d];
    col[i] = (_Float16)v;
}

// ConvLSTM gate update (keras order i,f,c,o; hard_sigmoid gates)
__global__ void gk_clstm_gates(const float* __restrict__ zg /* (B,1024,256) */,
                               float* c, float* h,
                               _Float16* __restrict__ hseq16 /* (B,P,C,D) */, int p)
{
    long i = (long)blockIdx.x * 256 + threadIdx.x;
    const long total = (long)cB * cC * cD;
    if (i >= total) return;
    long cell = i >> 6;
    int d = (int)(i & 63);
    const float* g = zg + cell * 256;
    float gi = g[d], gf = g[64 + d], gc = g[128 + d], go = g[192 + d];
    float fi = fminf(fmaxf(0.2f * gi + 0.5f, 0.f), 1.f);
    float ff = fminf(fmaxf(0.2f * gf + 0.5f, 0.f), 1.f);
    float fo = fminf(fmaxf(0.2f * go + 0.5f, 0.f), 1.f);
    float cn = ff * c[i] + fi * tanhf(gc);
    float hn = fo * tanhf(cn);
    c[i] = cn;
    h[i] = hn;
    long b = cell >> 10, cc = cell & 1023;
    hseq16[(((b * cP) + p) * cC + cc) * cD + d] = (_Float16)hn;
}

// deterministic hash-based uniform for gumbel noise
__device__ inline float gk_urand(unsigned a, unsigned b)
{
    unsigned x = a * 1664525u + b * 1013904223u + 0x9e3779b9u;
    x ^= x >> 16; x *= 0x7feb352du; x ^= x >> 15; x *= 0x846ca68bu; x ^= x >> 16;
    float u = (float)(x >> 8) * (1.f / 16777216.f);
    return fminf(fmaxf(u, 1e-6f), 1.f - 1e-6f);
}

// row-wise gumbel softmax over (N,C) -> f16
__global__ void gk_gumbel(const float* __restrict__ logits, _Float16* __restrict__ out16)
{
    __shared__ float red[256];
    int n = blockIdx.x, tid = threadIdx.x;
    float mx = -1e30f;
    for (int c = tid; c < cC; c += 256) {
        float g = -__logf(-__logf(gk_urand(n, c)));
        mx = fmaxf(mx, logits[(long)n * cC + c] + g);
    }
    red[tid] = mx; __syncthreads();
    for (int o = 128; o > 0; o >>= 1) {
        if (tid < o) red[tid] = fmaxf(red[tid], red[tid + o]);
        __syncthreads();
    }
    float rowmax = red[0]; __syncthreads();
    float sum = 0.f;
    for (int c = tid; c < cC; c += 256) {
        float g = -__logf(-__logf(gk_urand(n, c)));
        sum += __expf(logits[(long)n * cC + c] + g - rowmax);
    }
    red[tid] = sum; __syncthreads();
    for (int o = 128; o > 0; o >>= 1) {
        if (tid < o) red[tid] += red[tid + o];
        __syncthreads();
    }
    float inv = 1.f / red[0]; __syncthreads();
    for (int c = tid; c < cC; c += 256) {
        float g = -__logf(-__logf(gk_urand(n, c)));
        out16[(long)n * cC + c] =
            (_Float16)(__expf(logits[(long)n * cC + c] + g - rowmax) * inv);
    }
}

// pack [x | h] or [x | r*h] into (B,N,128) f32+f16 (GEMM operand)
__global__ void gk_pack_xh(const float* __restrict__ x, long xbs, long xoff,
                           const float* __restrict__ h,
                           const float* __restrict__ ru, // r = ru[row*128 + col-64]
                           float* __restrict__ out32, _Float16* __restrict__ out16)
{
    long i = (long)blockIdx.x * 256 + threadIdx.x;
    const long total = (long)cB * cN * 128;
    if (i >= total) return;
    long row = i >> 7;
    int col = (int)(i & 127);
    long b = row >> 10, n = row & 1023;
    float v;
    if (col < 64) {
        v = x[xoff + b * xbs + n * cD + col];
    } else {
        v = h[row * cD + (col - 64)];
        if (ru) v *= ru[row * 128 + (col - 64)];
    }
    out32[i] = v;
    out16[i] = (_Float16)v;
}

// GRU state update h = u*h + (1-u)*c ; optionally accumulate into Xf[:,p]
__global__ void gk_gru_h(const float* __restrict__ ru, const float* __restrict__ cand,
                         float* h, float* xf, long xfbs, long xfoff)
{
    long i = (long)blockIdx.x * 256 + threadIdx.x;
    const long total = (long)cB * cN * cD;
    if (i >= total) return;
    long row = i >> 6;
    int d = (int)(i & 63);
    float u = ru[row * 128 + 64 + d];
    float hn = u * h[i] + (1.f - u) * cand[i];
    h[i] = hn;
    if (xf) {
        long b = row >> 10, n = row & 1023;
        xf[xfoff + b * xfbs + n * cD + d] += hn;
    }
}

// (B,N,Q) -> (B,Q,N)
__global__ void gk_transpose_out(const float* __restrict__ Y, float* __restrict__ out)
{
    long i = (long)blockIdx.x * 256 + threadIdx.x;
    const long total = (long)cB * cQ * cN;
    if (i >= total) return;
    long n = i % cN;
    long bq = i / cN;
    long q = bq % cQ, b = bq / cQ;
    out[i] = Y[(b * cN + n) * cQ + q];
}

// ======================================================================
//  Host-side orchestration
// ======================================================================
static void gemm(hipStream_t s,
                 const _Float16* A, long sA, const _Float16* B, long sB,
                 const float* Cres, long sC, float beta, const float* bias,
                 float* D, long sD, _Float16* Dh, long sDh,
                 int M, int K, int Nc, int batch, float alpha, int act)
{
    dim3 grid((Nc + 63) / 64, (M + 127) / 128, batch);
    gk_wmma_gemm<<<grid, dim3(256), 0, s>>>(A, sA, B, sB, Cres, sC, beta, bias,
                                            D, sD, Dh, sDh, M, K, Nc, alpha, act);
}

static void cast16(hipStream_t s, const float* src, _Float16* dst, long n)
{
    gk_cast16<<<(int)((n + 255) / 256), 256, 0, s>>>(src, dst, n);
}

static void zerof(hipStream_t s, float* p, long n)
{
    gk_zero<<<(int)((n + 255) / 256), 256, 0, s>>>(p, n);
}

struct ChebBufs {
    const _Float16* sup16;
    float* x0_32; _Float16* x0_16;
    float* x1_32; _Float16* x1_16;
    float* x2_32; _Float16* x2_16;
    float* x3_32; _Float16* x3_16;
    float* ru;    float* cand;
};

// Chebyshev K=3 diffusion: x1 = S@x0; x2 = 2 S@x1 - x0; x3 = 2 S@x2 - x1
static void cheb_chain(hipStream_t s, const ChebBufs& b)
{
    const long sX = (long)cN * 128;
    gemm(s, b.sup16, 0, b.x0_16, sX, nullptr, 0, 0.f, nullptr,
         b.x1_32, sX, b.x1_16, sX, cN, cN, 128, cB, 1.f, 0);
    gemm(s, b.sup16, 0, b.x1_16, sX, b.x0_32, sX, -1.f, nullptr,
         b.x2_32, sX, b.x2_16, sX, cN, cN, 128, cB, 2.f, 0);
    gemm(s, b.sup16, 0, b.x2_16, sX, b.x1_32, sX, -1.f, nullptr,
         b.x3_32, sX, b.x3_16, sX, cN, cN, 128, cB, 2.f, 0);
}

// out = act( sum_j xj @ W[j*128:(j+1)*128, :Nout] + bias )  (no physical concat)
static void gconv_out(hipStream_t s, const ChebBufs& b, const _Float16* W,
                      const float* bias, int Nout, float* out, int act)
{
    const int M = cB * cN;
    const _Float16* xs[4] = { b.x0_16, b.x1_16, b.x2_16, b.x3_16 };
    for (int j = 0; j < 4; ++j) {
        bool last = (j == 3);
        gemm(s, xs[j], 0, W + (long)j * 128 * Nout, 0,
             j ? out : nullptr, 0, j ? 1.f : 0.f,
             last ? bias : nullptr, out, 0, nullptr, 0,
             M, 128, Nout, 1, 1.f, last ? act : 0);
    }
}

static void dcgru_step(hipStream_t s, const float* x, long xbs, long xoff,
                       float* h, const ChebBufs& b,
                       const _Float16* w_ru, const float* b_ru,
                       const _Float16* w_c, const float* b_c,
                       float* xf, long xfbs, long xfoff)
{
    const long tot128 = (long)cB * cN * 128;
    const long tot64  = (long)cB * cN * cD;
    int g128 = (int)((tot128 + 255) / 256);
    int g64  = (int)((tot64 + 255) / 256);

    gk_pack_xh<<<g128, 256, 0, s>>>(x, xbs, xoff, h, nullptr, b.x0_32, b.x0_16);
    cheb_chain(s, b);
    gconv_out(s, b, w_ru, b_ru, 128, b.ru, /*sigmoid*/2);

    gk_pack_xh<<<g128, 256, 0, s>>>(x, xbs, xoff, h, b.ru, b.x0_32, b.x0_16);
    cheb_chain(s, b);
    gconv_out(s, b, w_c, b_c, 64, b.cand, /*tanh*/3);

    gk_gru_h<<<g64, 256, 0, s>>>(b.ru, b.cand, h, xf, xfbs, xfoff);
}

extern "C" void kernel_launch(void* const* d_in, const int* in_sizes, int n_in,
                              void* d_out, int out_size, void* d_ws, size_t ws_size,
                              hipStream_t stream)
{
    (void)in_sizes; (void)n_in; (void)out_size; (void)ws_size;
    // ---- inputs in alphabetical pytree-flatten order ----
    const int*   TE       = (const int*)  d_in[0];   // (B,24,2)
    const float* X        = (const float*)d_in[1];   // (B,P,N)
    const float* Zin      = (const float*)d_in[2];   // (B,P,C,1)
    const float* adj      = (const float*)d_in[3];   // (N,N)
    const float* SE_X     = (const float*)d_in[4];   // (N,D)
    const float* SE_Z     = (const float*)d_in[5];   // (C,D)
    const float* cl_wh    = (const float*)d_in[6];   // (3,3,D,4D)
    const float* cl_wx    = (const float*)d_in[7];
    const float* dx_b_c   = (const float*)d_in[8];
    const float* dx_b_ru  = (const float*)d_in[9];
    const float* dx_w_c   = (const float*)d_in[10];  // (512,64)
    const float* dx_w_ru  = (const float*)d_in[11];  // (512,128)
    const float* dz_b_c   = (const float*)d_in[12];
    const float* dz_b_ru  = (const float*)d_in[13];
    const float* dz_w_c   = (const float*)d_in[14];
    const float* dz_w_ru  = (const float*)d_in[15];
    const float* fxi_b1   = (const float*)d_in[16];
    const float* fxi_b2   = (const float*)d_in[17];
    const float* fxi_w1   = (const float*)d_in[18];  // (1,64)
    const float* fxi_w2   = (const float*)d_in[19];  // (64,64)
    const float* fxo_b1   = (const float*)d_in[20];
    const float* fxo_b2   = (const float*)d_in[21];
    const float* fxo_w1   = (const float*)d_in[22];  // (64,64)
    const float* fxo_w2   = (const float*)d_in[23];  // (64,12)
    const float* fzi_b1   = (const float*)d_in[24];
    const float* fzi_b2   = (const float*)d_in[25];
    const float* fzi_w1   = (const float*)d_in[26];
    const float* fzi_w2   = (const float*)d_in[27];
    const float* movGR    = (const float*)d_in[28];  // (N,C)
    const float* sx_se_b1 = (const float*)d_in[29];
    const float* sx_se_b2 = (const float*)d_in[30];
    const float* sx_se_w1 = (const float*)d_in[31];
    const float* sx_se_w2 = (const float*)d_in[32];
    const float* sx_te_b1 = (const float*)d_in[33];
    const float* sx_te_b2 = (const float*)d_in[34];
    const float* sx_te_w1 = (const float*)d_in[35];  // (295,64)
    const float* sx_te_w2 = (const float*)d_in[36];
    const float* sz_se_b1 = (const float*)d_in[37];
    const float* sz_se_b2 = (const float*)d_in[38];
    const float* sz_se_w1 = (const float*)d_in[39];
    const float* sz_se_w2 = (const float*)d_in[40];
    const float* sz_te_b1 = (const float*)d_in[41];
    const float* sz_te_b2 = (const float*)d_in[42];
    const float* sz_te_w1 = (const float*)d_in[43];
    const float* sz_te_w2 = (const float*)d_in[44];
    float* out = (float*)d_out;

    // ---- workspace arena ----
    char* wp = (char*)d_ws;
    auto alloc = [&](size_t bytes) -> char* {
        char* r = wp;
        wp += (bytes + 255) & ~(size_t)255;
        return r;
    };
    auto af = [&](long n) { return (float*)    alloc((size_t)n * 4); };
    auto ah = [&](long n) { return (_Float16*) alloc((size_t)n * 2); };

    float*    deg      = af(cN);
    _Float16* sup16    = ah((long)cN * cN);
    _Float16* adjGR16  = ah((long)cN * cC);
    _Float16* SEX16    = ah((long)cN * cD);
    _Float16* SEZ16    = ah((long)cC * cD);
    _Float16* sxw1h    = ah(64 * 64);
    _Float16* sxw2h    = ah(64 * 64);
    _Float16* szw1h    = ah(64 * 64);
    _Float16* szw2h    = ah(64 * 64);
    _Float16* fxow1h   = ah(64 * 64);
    _Float16* fxow2h   = ah(64 * 12);
    _Float16* clwx16   = ah((long)576 * 256);
    _Float16* clwh16   = ah((long)576 * 256);
    _Float16* wruZ16   = ah((long)512 * 128);
    _Float16* wcZ16    = ah((long)512 * 64);
    _Float16* wruX16   = ah((long)512 * 128);
    _Float16* wcX16    = ah((long)512 * 64);
    float*    teX      = af((long)cB * cP * cD);
    float*    teZ      = af((long)cB * cP * cD);
    float*    seHid32  = af((long)cN * cD);
    _Float16* seHid16  = ah((long)cN * cD);
    float*    seX32    = af((long)cN * cD);
    float*    seZ32    = af((long)cC * cD);
    float*    Zf       = af((long)cB * cP * cC * cD);   // clstm input seq
    _Float16* Zfo16    = ah((long)cB * cP * cC * cD);   // clstm output seq (f16)
    float*    cl_c     = af((long)cB * cC * cD);
    float*    cl_h     = af((long)cB * cC * cD);
    _Float16* colbuf   = ah((long)cB * 1024 * 576);
    float*    zg       = af((long)cB * 1024 * 256);
    float*    Zn       = af((long)cB * cN * cD);
    float*    Xf       = af((long)cB * cP * cN * cD);
    float*    x0_32    = af((long)cB * cN * 128);
    _Float16* x0_16    = ah((long)cB * cN * 128);
    float*    x1_32    = af((long)cB * cN * 128);
    _Float16* x1_16    = ah((long)cB * cN * 128);
    float*    x2_32    = af((long)cB * cN * 128);
    _Float16* x2_16    = ah((long)cB * cN * 128);
    float*    x3_32    = af((long)cB * cN * 128);
    _Float16* x3_16    = ah((long)cB * cN * 128);
    float*    ru       = af((long)cB * cN * 128);
    float*    cand     = af((long)cB * cN * cD);
    float*    h_z      = af((long)cB * cN * cD);
    float*    h_x      = af((long)cB * cN * cD);
    _Float16* hX16     = ah((long)cB * cN * cD);
    float*    fcHid32  = af((long)cB * cN * cD);
    _Float16* fcHid16  = ah((long)cB * cN * cD);
    float*    Yb       = af((long)cB * cN * cQ);

    // ---- scaled Laplacian support (f16, stays in L2) ----
    gk_degree<<<cN, 256, 0, stream>>>(adj, deg);
    gk_support<<<(int)(((long)cN * cN + 255) / 256), 256, 0, stream>>>(adj, deg, sup16);

    // ---- weight & SE casts ----
    cast16(stream, sx_se_w1, sxw1h, 64 * 64);
    cast16(stream, sx_se_w2, sxw2h, 64 * 64);
    cast16(stream, sz_se_w1, szw1h, 64 * 64);
    cast16(stream, sz_se_w2, szw2h, 64 * 64);
    cast16(stream, fxo_w1, fxow1h, 64 * 64);
    cast16(stream, fxo_w2, fxow2h, 64 * 12);
    cast16(stream, cl_wx, clwx16, (long)576 * 256);
    cast16(stream, cl_wh, clwh16, (long)576 * 256);
    cast16(stream, dz_w_ru, wruZ16, (long)512 * 128);
    cast16(stream, dz_w_c,  wcZ16,  (long)512 * 64);
    cast16(stream, dx_w_ru, wruX16, (long)512 * 128);
    cast16(stream, dx_w_c,  wcX16,  (long)512 * 64);
    cast16(stream, SE_X, SEX16, (long)cN * cD);
    cast16(stream, SE_Z, SEZ16, (long)cC * cD);

    // ---- ST embeddings ----
    gk_te_embed<<<cB * cP, cD, 0, stream>>>(TE, sx_te_w1, sx_te_b1, sx_te_w2, sx_te_b2, teX);
    gk_te_embed<<<cB * cP, cD, 0, stream>>>(TE, sz_te_w1, sz_te_b1, sz_te_w2, sz_te_b2, teZ);
    // se = fc2(se_mlp, SE)
    gemm(stream, SEX16, 0, sxw1h, 0, nullptr, 0, 0.f, sx_se_b1,
         seHid32, 0, seHid16, 0, cN, 64, 64, 1, 1.f, /*relu*/1);
    gemm(stream, seHid16, 0, sxw2h, 0, nullptr, 0, 0.f, sx_se_b2,
         seX32, 0, nullptr, 0, cN, 64, 64, 1, 1.f, 0);
    gemm(stream, SEZ16, 0, szw1h, 0, nullptr, 0, 0.f, sz_se_b1,
         seHid32, 0, seHid16, 0, cC, 64, 64, 1, 1.f, 1);
    gemm(stream, seHid16, 0, szw2h, 0, nullptr, 0, 0.f, sz_se_b2,
         seZ32, 0, nullptr, 0, cC, 64, 64, 1, 1.f, 0);

    // ---- input embeds: Zf = fc2(Z)+STEZ ; Xf = fc2(X)+STEX (ZX added later) ----
    gk_in1_embed<<<cB * cP * cC, cD, 0, stream>>>(Zin, fzi_w1, fzi_b1, fzi_w2, fzi_b2,
                                                  seZ32, teZ, Zf);
    gk_in1_embed<<<cB * cP * cN, cD, 0, stream>>>(X, fxi_w1, fxi_b1, fxi_w2, fxi_b2,
                                                  seX32, teX, Xf);

    // ---- ConvLSTM over cell branch (im2col + WMMA GEMMs) ----
    zerof(stream, cl_h, (long)cB * cC * cD);
    zerof(stream, cl_c, (long)cB * cC * cD);
    {
        const long colTot = (long)cB * 1024 * 576;
        int gcol = (int)((colTot + 255) / 256);
        const long gateTot = (long)cB * cC * cD;
        int ggate = (int)((gateTot + 255) / 256);
        for (int p = 0; p < cP; ++p) {
            gk_im2col<<<gcol, 256, 0, stream>>>(Zf, (long)cP * cC * cD,
                                                (long)p * cC * cD, colbuf);
            gemm(stream, colbuf, 0, clwx16, 0, nullptr, 0, 0.f, nullptr,
                 zg, 0, nullptr, 0, cB * 1024, 576, 256, 1, 1.f, 0);
            gk_im2col<<<gcol, 256, 0, stream>>>(cl_h, (long)cC * cD, 0, colbuf);
            gemm(stream, colbuf, 0, clwh16, 0, zg, 0, 1.f, nullptr,
                 zg, 0, nullptr, 0, cB * 1024, 576, 256, 1, 1.f, 0);
            gk_clstm_gates<<<ggate, 256, 0, stream>>>(zg, cl_c, cl_h, Zfo16, p);
        }
    }

    // ---- gumbel-softmax cell->node assignment ----
    gk_gumbel<<<cN, 256, 0, stream>>>(movGR, adjGR16);

    ChebBufs cb = { sup16, x0_32, x0_16, x1_32, x1_16, x2_32, x2_16,
                    x3_32, x3_16, ru, cand };

    // ---- DCGRU (cell->node branch), accumulates ZX into Xf per step ----
    zerof(stream, h_z, (long)cB * cN * cD);
    for (int p = 0; p < cP; ++p) {
        // Zn[b] = adj_GR @ Zf_out[b,p]
        gemm(stream, adjGR16, 0, Zfo16 + (long)p * cC * cD, (long)cP * cC * cD,
             nullptr, 0, 0.f, nullptr, Zn, (long)cN * cD, nullptr, 0,
             cN, cC, cD, cB, 1.f, 0);
        dcgru_step(stream, Zn, (long)cN * cD, 0, h_z, cb,
                   wruZ16, dz_b_ru, wcZ16, dz_b_c,
                   Xf, (long)cP * cN * cD, (long)p * cN * cD);
    }

    // ---- DCGRU (node branch), keep final state ----
    zerof(stream, h_x, (long)cB * cN * cD);
    for (int p = 0; p < cP; ++p) {
        dcgru_step(stream, Xf, (long)cP * cN * cD, (long)p * cN * cD, h_x, cb,
                   wruX16, dx_b_ru, wcX16, dx_b_c,
                   nullptr, 0, 0);
    }

    // ---- output head: Y = fc2(fc_x_out, hX); transpose to (B,Q,N) ----
    cast16(stream, h_x, hX16, (long)cB * cN * cD);
    gemm(stream, hX16, 0, fxow1h, 0, nullptr, 0, 0.f, fxo_b1,
         fcHid32, 0, fcHid16, 0, cB * cN, 64, 64, 1, 1.f, /*relu*/1);
    gemm(stream, fcHid16, 0, fxow2h, 0, nullptr, 0, 0.f, fxo_b2,
         Yb, 0, nullptr, 0, cB * cN, 64, cQ, 1, 1.f, 0);
    gk_transpose_out<<<(int)(((long)cB * cQ * cN + 255) / 256), 256, 0, stream>>>(Yb, out);
}
